// PatternGrowth_16638703305224
// MI455X (gfx1250) — matrix operand order repaired and checked by reading, not compiled, not run on hardware
//
#include <hip/hip_runtime.h>
#include <math.h>

typedef _Float16 h16;
typedef __attribute__((ext_vector_type(16))) _Float16 v16h;
typedef __attribute__((ext_vector_type(8)))  _Float16 v8h;
typedef __attribute__((ext_vector_type(8)))  float    v8f;

#define EPSBN 1e-5f

__device__ __forceinline__ float fsigmoid(float z) { return 1.0f / (1.0f + __expf(-z)); }

#define SHUF16(lo, hi) __builtin_shufflevector((lo), (hi), 0,1,2,3,4,5,6,7,8,9,10,11,12,13,14,15)

__device__ __forceinline__ v8h zero8h() {
  v8h z = {(h16)0,(h16)0,(h16)0,(h16)0,(h16)0,(h16)0,(h16)0,(h16)0};
  return z;
}

// ---------------------------------------------------------------------------
// K1/K10: channel moments of a (4,3,npix) tensor: sums (3) + upper-tri 2nd moments (6)
// ---------------------------------------------------------------------------
__global__ __launch_bounds__(256) void k_mom3(const float* __restrict__ src, int npix,
                                              float* sum3, float* m6) {
  int tid = threadIdx.x;
  float s0=0,s1=0,s2=0,m00=0,m01=0,m02=0,m11=0,m12=0,m22=0;
  int total = 4 * npix;
  for (int i = blockIdx.x * blockDim.x + tid; i < total; i += gridDim.x * blockDim.x) {
    int b = i / npix, p = i - b * npix;
    const float* base = src + (size_t)(b * 3) * npix + p;
    float v0 = base[0], v1 = base[npix], v2 = base[2 * npix];
    s0 += v0; s1 += v1; s2 += v2;
    m00 += v0*v0; m01 += v0*v1; m02 += v0*v2;
    m11 += v1*v1; m12 += v1*v2; m22 += v2*v2;
  }
  __shared__ float red[256];
  float vals[9] = {s0,s1,s2,m00,m01,m02,m11,m12,m22};
  for (int q = 0; q < 9; ++q) {
    red[tid] = vals[q];
    __syncthreads();
    for (int st = 128; st > 0; st >>= 1) {
      if (tid < st) red[tid] += red[tid + st];
      __syncthreads();
    }
    if (tid == 0) {
      if (q < 3) atomicAdd(&sum3[q], red[0]);
      else       atomicAdd(&m6[q - 3], red[0]);
    }
    __syncthreads();
  }
}

// ---------------------------------------------------------------------------
// K2: analytic BN coefficients for compenv (ce) and cell_div stage 1 (cd)
// ---------------------------------------------------------------------------
__global__ void k2_coeffs1(const float* stats,
                           const float* ce_w1, const float* ce_g, const float* ce_b,
                           const float* cd_wt, const float* cd_g, const float* cd_b,
                           float* aCE, float* cCE, float* waCD, float* cCD) {
  int o = threadIdx.x;
  if (o >= 48) return;
  const float inv = 1.0f / (float)(4 * 256 * 256);
  float mean[3] = {stats[0]*inv, stats[1]*inv, stats[2]*inv};
  float S00 = stats[3]*inv, S01 = stats[4]*inv, S02 = stats[5]*inv;
  float S11 = stats[6]*inv, S12 = stats[7]*inv, S22 = stats[8]*inv;
  float Sd[3] = {S00, S11, S22};
  float w0 = ce_w1[o*3], w1 = ce_w1[o*3+1], w2 = ce_w1[o*3+2];
  float mu = w0*mean[0] + w1*mean[1] + w2*mean[2];
  float e2 = w0*w0*S00 + w1*w1*S11 + w2*w2*S22
           + 2.0f*(w0*w1*S01 + w0*w2*S02 + w1*w2*S12);
  float A = ce_g[o] * rsqrtf(e2 - mu*mu + EPSBN);
  aCE[o] = A;
  cCE[o] = ce_b[o] - A * mu;
  int g = o / 16, op = o % 16;
  float wt4[4], sw = 0.0f, sw2 = 0.0f;
  #pragma unroll
  for (int t = 0; t < 4; ++t) {
    wt4[t] = cd_wt[(g*16 + op)*4 + t];
    sw += wt4[t]; sw2 += wt4[t]*wt4[t];
  }
  float muc = 0.25f * sw * mean[g];
  float e2c = 0.25f * sw2 * Sd[g];
  float Ac = cd_g[o] * rsqrtf(e2c - muc*muc + EPSBN);
  cCD[o] = cd_b[o] - Ac * muc;
  #pragma unroll
  for (int t = 0; t < 4; ++t) waCD[o*4 + t] = Ac * wt4[t];
}

// ---------------------------------------------------------------------------
// K3: compenv: 8 rolled dirs, fused 1x1 -> BN -> tanh -> 1x1. ENV (B,8,256,256)
// ---------------------------------------------------------------------------
__global__ __launch_bounds__(256) void k3_compenv(const float* __restrict__ x,
                                                  const float* __restrict__ w1,
                                                  const float* __restrict__ w2,
                                                  const float* __restrict__ aCE,
                                                  const float* __restrict__ cCE,
                                                  float* __restrict__ ENVo) {
  __shared__ float sw1[144], sw2[48], sa[48], sc[48];
  int tid = threadIdx.x;
  if (tid < 144) sw1[tid] = w1[tid];
  if (tid < 48) { sw2[tid] = w2[tid]; sa[tid] = aCE[tid]; sc[tid] = cCE[tid]; }
  __syncthreads();
  int gid = blockIdx.x * 256 + tid;
  if (gid >= 4 * 65536) return;
  int b = gid >> 16, p = gid & 65535, y = p >> 8, xo = p & 255;
  const int TX[8] = {0, 1, 1, 1, 0, -1, -1, -1};
  const int TY[8] = {1, 1, 0, -1, -1, -1, 0, 1};
  for (int k = 0; k < 8; ++k) {
    int yy = (y + TY[k]) & 255;
    int xx = (xo - TX[k]) & 255;
    const float* xb = x + ((size_t)(b*3))*65536 + yy*256 + xx;
    float x0 = xb[0], x1 = xb[65536], x2 = xb[2*65536];
    float e = 0.0f;
    for (int o = 0; o < 48; ++o) {
      float yp = sw1[o*3]*x0 + sw1[o*3+1]*x1 + sw1[o*3+2]*x2;
      e += sw2[o] * tanhf(sa[o]*yp + sc[o]);
    }
    ENVo[((b*8 + k)*256 + y)*256 + xo] = e;
  }
}

// ---------------------------------------------------------------------------
// K4: dd convT1 (11->48, 3x3, full conv pad 2) + tanh -> h1dd f16 (B,48,258,258)
// ---------------------------------------------------------------------------
__global__ __launch_bounds__(128) void k4_ddconvT1(const float* __restrict__ x,
                                                   const float* __restrict__ ENV,
                                                   const float* __restrict__ wt, // (11,48,3,3)
                                                   h16* __restrict__ h1) {
  __shared__ float sW[48 * 99]; // [o][i*9+tap], spatially flipped
  int tid = threadIdx.x;
  for (int idx = tid; idx < 48 * 99; idx += 128) {
    int o = idx / 99, r = idx % 99, i = r / 9, tap = r % 9;
    int ty = tap / 3, tx = tap % 3;
    sW[idx] = wt[(i*48 + o)*9 + (2 - ty)*3 + (2 - tx)];
  }
  __syncthreads();
  int gid = blockIdx.x * 128 + tid;
  if (gid >= 4 * 258 * 258) return;
  int b = gid / (258*258), r = gid % (258*258), y = r / 258, xo = r % 258;
  float acc[48];
  #pragma unroll
  for (int o = 0; o < 48; ++o) acc[o] = 0.0f;
  for (int i = 0; i < 11; ++i) {
    for (int tap = 0; tap < 9; ++tap) {
      int yy = y - 2 + tap/3, xx = xo - 2 + tap%3;
      float v = 0.0f;
      if (yy >= 0 && yy < 256 && xx >= 0 && xx < 256)
        v = (i < 3) ? x[((b*3 + i)*256 + yy)*256 + xx]
                    : ENV[((b*8 + (i-3))*256 + yy)*256 + xx];
      const float* w = &sW[i*9 + tap];
      #pragma unroll
      for (int o = 0; o < 48; ++o) acc[o] += w[o*99] * v;
    }
  }
  #pragma unroll
  for (int o = 0; o < 48; ++o)
    h1[((b*48 + o)*258 + y)*258 + xo] = (h16)tanhf(acc[o]);
}

// ---------------------------------------------------------------------------
// K5: dd conv2 48->48 3x3 VALID, WMMA implicit GEMM, channel-fastest LDS layout:
//   patch sP[row(10)][col(18)][c(48)] f16, weights sW[tap(9)][o(48)][c(48)] f16.
// K ordered (tap, c): per tap 2 chunks of 32 (c 0..31, c 32..47 + 16 compile-time
// zero A elements). Every fragment = 1-2 aligned ds_load_b128.
// ---------------------------------------------------------------------------
__global__ __launch_bounds__(128) void k5_ddconv2(const h16* __restrict__ h1,
                                                  const float* __restrict__ wc, // (48,48,3,3)
                                                  h16* __restrict__ h2) {
  __shared__ h16 sP[10 * 18 * 48];       // 17280 B
  __shared__ h16 sW[9 * 48 * 48 + 32];   // 41536 B (slack row zeroed)
  const int tid = threadIdx.x;
  const int b = blockIdx.z;
  const int ty0 = blockIdx.y * 8;
  const int tx0 = blockIdx.x * 16;
  // weights: wc[o][c][tap] -> sW[tap][o][c]
  for (int i = tid; i < 48 * 48 * 9; i += 128) {
    int o = i / 432, r = i % 432, c = r / 9, tap = r % 9;
    sW[(tap*48 + o)*48 + c] = (h16)wc[i];
  }
  if (tid < 32) sW[9*48*48 + tid] = (h16)0.0f;
  // patch: coalesced on col, channel-fastest in LDS
  for (int i = tid; i < 48 * 180; i += 128) {
    int c = i / 180, r = i % 180, row = r / 18, col = r % 18;
    sP[(row*18 + col)*48 + c] = h1[((b*48 + c)*258 + (ty0 + row))*258 + (tx0 + col)];
  }
  __syncthreads();
  const int wave = tid >> 5, lane = tid & 31;
  const int lm = lane & 15, half = lane >> 4;
  v8f acc[2][3];
  #pragma unroll
  for (int mt = 0; mt < 2; ++mt)
    #pragma unroll
    for (int nt = 0; nt < 3; ++nt) acc[mt][nt] = (v8f){0,0,0,0,0,0,0,0};
  #pragma unroll
  for (int tap = 0; tap < 9; ++tap) {
    const int ty = tap / 3, tx = tap % 3;
    #pragma unroll
    for (int cb = 0; cb < 2; ++cb) {
      v16h a[2];
      #pragma unroll
      for (int mt = 0; mt < 2; ++mt) {
        int py = wave + mt * 4;
        const h16* ab = &sP[((py + ty)*18 + (lm + tx))*48 + cb*32 + half*8];
        v8h lo = *(const v8h*)ab;
        v8h hi = (cb == 0) ? *(const v8h*)(ab + 16) : zero8h();
        a[mt] = SHUF16(lo, hi);
      }
      #pragma unroll
      for (int nt = 0; nt < 3; ++nt) {
        // B row (tap, n); chunk-B half=1 overreads neighbor weights: finite,
        // multiplied by compile-time-zero A elements.
        const h16* bb = &sW[(tap*48 + nt*16 + lm)*48 + cb*32 + half*8];
        v8h blo = *(const v8h*)bb;
        v8h bhi = *(const v8h*)(bb + 16);
        v16h bf = SHUF16(blo, bhi);
        #pragma unroll
        for (int mt = 0; mt < 2; ++mt)
          acc[mt][nt] = __builtin_amdgcn_wmma_f32_16x16x32_f16(
              false, a[mt], false, bf, (short)0, acc[mt][nt], false, false);
      }
    }
  }
  #pragma unroll
  for (int mt = 0; mt < 2; ++mt) {
    int y = ty0 + wave + mt * 4;
    #pragma unroll
    for (int nt = 0; nt < 3; ++nt) {
      int oc = nt*16 + lm;
      #pragma unroll
      for (int r = 0; r < 8; ++r) {
        int xcoord = tx0 + r + 8*half;
        h2[((b*48 + oc)*256 + y)*256 + xcoord] = (h16)tanhf(acc[mt][nt][r]);
      }
    }
  }
}

// ---------------------------------------------------------------------------
// K6: per-channel moments of h2dd (48 channels)
// ---------------------------------------------------------------------------
__global__ __launch_bounds__(256) void k6_ddmom(const h16* __restrict__ h2,
                                                float* ddsum, float* ddsum2) {
  int c = blockIdx.x, tid = threadIdx.x;
  float s = 0.0f, s2 = 0.0f;
  for (int i = blockIdx.y*256 + tid; i < 4 * 65536; i += gridDim.y * 256) {
    int b = i >> 16, p = i & 65535;
    float v = (float)h2[(size_t)(b*48 + c)*65536 + p];
    s += v; s2 += v*v;
  }
  __shared__ float r1[256], r2[256];
  r1[tid] = s; r2[tid] = s2;
  __syncthreads();
  for (int st = 128; st > 0; st >>= 1) {
    if (tid < st) { r1[tid] += r1[tid+st]; r2[tid] += r2[tid+st]; }
    __syncthreads();
  }
  if (tid == 0) { atomicAdd(&ddsum[c], r1[0]); atomicAdd(&ddsum2[c], r2[0]); }
}

// ---------------------------------------------------------------------------
// K8: cd funnel convT (grouped 16->16, 3x3 full conv pad 2) WMMA. Stage-1
// (convT s2 + BN + tanh) computed analytically during LDS patch staging.
// K ordered (tap, c16): chunk = tap pair. sP[row][col][c16], sW[tap(10)][o16][c16]
// (tap 9 row zeroed). Fragments = aligned ds_load_b128 pairs.
// ---------------------------------------------------------------------------
__global__ __launch_bounds__(128) void k8_cdconv2(const float* __restrict__ x,
                                                  const float* __restrict__ wt,   // cd_f_wt (48,16,3,3)
                                                  const float* __restrict__ waCD, // [48][4]
                                                  const float* __restrict__ cCD,  // [48]
                                                  h16* __restrict__ h2c) {
  __shared__ h16 sP[10 * 18 * 16];    // 5760 B
  __shared__ h16 sW[10 * 16 * 16];    // 5120 B (tap 9 = zero pad)
  __shared__ float sWA[64], sC[16];
  int tid = threadIdx.x;
  int bz = blockIdx.z, b = bz / 3, g = bz % 3;
  int ty0 = blockIdx.y * 8, tx0 = blockIdx.x * 16;
  if (tid < 64) sWA[tid] = waCD[g*64 + tid];
  if (tid < 16) sC[tid] = cCD[g*16 + tid];
  // weights: sW[tap][o][c] = flip(cd_f_wt)
  for (int i = tid; i < 16 * 16 * 9; i += 128) {
    int c = i / 144, r = i % 144, o = r / 9, tap = r % 9;
    int ty = tap / 3, tx = tap % 3;
    sW[(tap*16 + o)*16 + c] = (h16)wt[((g*16 + c)*16 + o)*9 + (2 - ty)*3 + (2 - tx)];
  }
  for (int i = tid; i < 256; i += 128) sW[9*16*16 + i] = (h16)0.0f;
  __syncthreads();
  // patch with on-the-fly stage-1 (analytic BN + tanh), channel-fastest
  for (int i = tid; i < 16 * 180; i += 128) {
    int c = i / 180, r = i % 180, row = r / 18, col = r % 18;
    int Y = ty0 - 2 + row, X = tx0 - 2 + col;
    float v = 0.0f;
    if (Y >= 0 && Y < 512 && X >= 0 && X < 512) {
      float xv = x[((b*3 + g)*256 + (Y >> 1))*256 + (X >> 1)];
      int tap = ((Y & 1) << 1) | (X & 1);
      v = tanhf(sWA[c*4 + tap]*xv + sC[c]);
    }
    sP[(row*18 + col)*16 + c] = (h16)v;
  }
  __syncthreads();
  int wave = tid >> 5, lane = tid & 31, lm = lane & 15, half = lane >> 4;
  v8f acc[2];
  acc[0] = (v8f){0,0,0,0,0,0,0,0};
  acc[1] = (v8f){0,0,0,0,0,0,0,0};
  #pragma unroll
  for (int q = 0; q < 5; ++q) {
    const int t0 = 2*q, t1 = 2*q + 1;          // t1==9 for q==4 (pad)
    const int ty0q = t0/3, tx0q = t0%3;
    const int ty1q = t1/3, tx1q = t1%3;        // unused when q==4
    v16h a[2];
    #pragma unroll
    for (int mt = 0; mt < 2; ++mt) {
      int py = wave + mt * 4;
      const h16* a0 = &sP[((py + ty0q)*18 + (lm + tx0q))*16 + half*8];
      v8h lo = *(const v8h*)a0;
      v8h hi;
      if (q < 4) {
        const h16* a1 = &sP[((py + ty1q)*18 + (lm + tx1q))*16 + half*8];
        hi = *(const v8h*)a1;
      } else {
        hi = zero8h();
      }
      a[mt] = SHUF16(lo, hi);
    }
    int tsel = t0;                              // tap 9 row is zeroed
    const h16* bb = &sW[(tsel*16 + lm)*16 + half*8];
    v8h blo = *(const v8h*)bb;
    v8h bhi = *(const v8h*)(bb + 256);
    v16h bf = SHUF16(blo, bhi);
    #pragma unroll
    for (int mt = 0; mt < 2; ++mt)
      acc[mt] = __builtin_amdgcn_wmma_f32_16x16x32_f16(
          false, a[mt], false, bf, (short)0, acc[mt], false, false);
  }
  #pragma unroll
  for (int mt = 0; mt < 2; ++mt) {
    int y = ty0 + wave + mt * 4;
    if (y >= 514) continue;
    #pragma unroll
    for (int r = 0; r < 8; ++r) {
      int xcoord = tx0 + r + 8*half;
      if (xcoord < 514)
        h2c[((size_t)(b*48 + g*16 + lm)*514 + y)*514 + xcoord] = (h16)tanhf(acc[mt][r]);
    }
  }
}

// ---------------------------------------------------------------------------
// K9: xg = tanh(grouped conv 16->1, 3x3 VALID) on h2cd -> (B,3,512,512) f32
// ---------------------------------------------------------------------------
__global__ __launch_bounds__(256) void k9_xg(const h16* __restrict__ h2c,
                                             const float* __restrict__ wc, // (3,16,3,3)
                                             float* __restrict__ xg) {
  __shared__ float sw[432];
  int tid = threadIdx.x;
  for (int i = tid; i < 432; i += 256) sw[i] = wc[i];
  __syncthreads();
  int gid = blockIdx.x * 256 + tid;
  if (gid >= 4 * 512 * 512) return;
  int b = gid / (512*512), p = gid % (512*512), Y = p / 512, X = p % 512;
  for (int g = 0; g < 3; ++g) {
    float acc = 0.0f;
    for (int c = 0; c < 16; ++c) {
      const h16* base = &h2c[((size_t)(b*48 + g*16 + c)*514 + Y)*514 + X];
      const float* w = &sw[(g*16 + c)*9];
      #pragma unroll
      for (int t = 0; t < 3; ++t)
        #pragma unroll
        for (int u = 0; u < 3; ++u)
          acc += w[t*3 + u] * (float)base[t*514 + u];
    }
    xg[((size_t)(b*3 + g)*512 + Y)*512 + X] = tanhf(acc);
  }
}

// ---------------------------------------------------------------------------
// K11: coefficients for dd BN (from h2dd moments) and born BN (from xg moments)
// ---------------------------------------------------------------------------
__global__ void k11_coeffs2(const float* stats,
                            const float* dd_g, const float* dd_b,
                            const float* bw1, const float* bg, const float* bb,
                            float* aDD, float* cDD, float* aB, float* cB) {
  int o = threadIdx.x;
  if (o >= 48) return;
  const float invd = 1.0f / (float)(4 * 256 * 256);
  float mu = stats[9 + o] * invd;
  float var = stats[57 + o] * invd - mu*mu;
  float A = dd_g[o] * rsqrtf(var + EPSBN);
  aDD[o] = A; cDD[o] = dd_b[o] - A * mu;
  const float invg = 1.0f / (float)(4 * 512 * 512);
  float mean[3] = {stats[105]*invg, stats[106]*invg, stats[107]*invg};
  float S00 = stats[108]*invg, S01 = stats[109]*invg, S02 = stats[110]*invg;
  float S11 = stats[111]*invg, S12 = stats[112]*invg, S22 = stats[113]*invg;
  float w0 = bw1[o], w1 = bw1[48 + o], w2 = bw1[96 + o];
  float mub = w0*mean[0] + w1*mean[1] + w2*mean[2];
  float e2 = w0*w0*S00 + w1*w1*S11 + w2*w2*S22
           + 2.0f*(w0*w1*S01 + w0*w2*S02 + w1*w2*S12);
  float Ab = bg[o] * rsqrtf(e2 - mub*mub + EPSBN);
  aB[o] = Ab; cB[o] = bb[o] - Ab * mub;
}

// ---------------------------------------------------------------------------
// K12: fused CHANGE (BN+tanh+convT s2+sigmoid) + born (1x1-BN-tanh-1x1) + blend
// ---------------------------------------------------------------------------
__global__ __launch_bounds__(256) void k12_final(const h16* __restrict__ h2dd,
                                                 const float* __restrict__ xg,
                                                 const float* __restrict__ ddwt, // (48,1,2,2)
                                                 const float* __restrict__ aDD,
                                                 const float* __restrict__ cDD,
                                                 const float* __restrict__ bw1,
                                                 const float* __restrict__ bw2,
                                                 const float* __restrict__ aB,
                                                 const float* __restrict__ cB,
                                                 float* __restrict__ out) {
  __shared__ float sdw[192], sa[48], sc[48], sw1[144], sw2[144], sab[48], scb[48];
  int tid = threadIdx.x;
  if (tid < 192) sdw[tid] = ddwt[tid];
  if (tid < 144) { sw1[tid] = bw1[tid]; sw2[tid] = bw2[tid]; }
  if (tid < 48) { sa[tid] = aDD[tid]; sc[tid] = cDD[tid]; sab[tid] = aB[tid]; scb[tid] = cB[tid]; }
  __syncthreads();
  int gid = blockIdx.x * 256 + tid;
  if (gid >= 4 * 65536) return;
  int b = gid >> 16, p = gid & 65535, y = p >> 8, x = p & 255;
  const h16* hp = h2dd + (size_t)(b*48)*65536 + p;
  float h3[48];
  #pragma unroll
  for (int c = 0; c < 48; ++c)
    h3[c] = tanhf(sa[c]*(float)hp[(size_t)c*65536] + sc[c]);
  #pragma unroll
  for (int py = 0; py < 2; ++py)
    #pragma unroll
    for (int px = 0; px < 2; ++px) {
      int tap = py*2 + px;
      float z = 0.0f;
      for (int c = 0; c < 48; ++c) z += h3[c] * sdw[c*4 + tap];
      float ch = fsigmoid(z);
      int Y = 2*y + py, X = 2*x + px;
      const float* xgb = xg + (size_t)(b*3)*262144 + Y*512 + X;
      float g0 = xgb[0], g1 = xgb[262144], g2 = xgb[2*262144];
      float b0 = 0.0f, b1 = 0.0f, b2 = 0.0f;
      for (int o = 0; o < 48; ++o) {
        float yp = g0*sw1[o] + g1*sw1[48 + o] + g2*sw1[96 + o];
        float hb = tanhf(sab[o]*yp + scb[o]);
        b0 += hb*sw2[o*3]; b1 += hb*sw2[o*3 + 1]; b2 += hb*sw2[o*3 + 2];
      }
      float* ob = out + (size_t)(b*3)*262144 + Y*512 + X;
      ob[0]        = g0*(1.0f - ch) + ch*b0;
      ob[262144]   = g1*(1.0f - ch) + ch*b1;
      ob[2*262144] = g2*(1.0f - ch) + ch*b2;
    }
}

// ---------------------------------------------------------------------------
// Host launcher
// ---------------------------------------------------------------------------
extern "C" void kernel_launch(void* const* d_in, const int* in_sizes, int n_in,
                              void* d_out, int out_size, void* d_ws, size_t ws_size,
                              hipStream_t stream) {
  const float* x       = (const float*)d_in[0];
  const float* ce_w1   = (const float*)d_in[1];
  const float* ce_g    = (const float*)d_in[2];
  const float* ce_b    = (const float*)d_in[3];
  const float* ce_w2   = (const float*)d_in[4];
  const float* cd_wt   = (const float*)d_in[5];
  const float* cd_g    = (const float*)d_in[6];
  const float* cd_b    = (const float*)d_in[7];
  const float* cd_f_wt = (const float*)d_in[8];
  const float* cd_f_wc = (const float*)d_in[9];
  const float* born_w1 = (const float*)d_in[10];
  const float* born_g  = (const float*)d_in[11];
  const float* born_b  = (const float*)d_in[12];
  const float* born_w2 = (const float*)d_in[13];
  const float* dd_f_wt = (const float*)d_in[14];
  const float* dd_f_wc = (const float*)d_in[15];
  const float* dd_g    = (const float*)d_in[16];
  const float* dd_b    = (const float*)d_in[17];
  const float* dd_wt   = (const float*)d_in[18];
  float* out = (float*)d_out;

  const size_t OFF_ENV   = 0;
  const size_t OFF_H1DD  = OFF_ENV  + (size_t)8*262144*4;
  const size_t OFF_H2DD  = OFF_H1DD + (size_t)4*48*258*258*2;
  const size_t OFF_H2CD  = OFF_H2DD + (size_t)4*48*65536*2;
  const size_t OFF_XG    = OFF_H2CD + (size_t)4*48*514*514*2;
  const size_t OFF_STATS = OFF_XG   + (size_t)4*3*262144*4;
  const size_t NEEDED    = OFF_STATS + 4096;
  if (ws_size < NEEDED) return;

  char* ws = (char*)d_ws;
  float* ENV   = (float*)(ws + OFF_ENV);
  h16*   h1dd  = (h16*)(ws + OFF_H1DD);
  h16*   h2dd  = (h16*)(ws + OFF_H2DD);
  h16*   h2cd  = (h16*)(ws + OFF_H2CD);
  float* xg    = (float*)(ws + OFF_XG);
  float* stats = (float*)(ws + OFF_STATS);
  float* xsum  = stats + 0;   float* xm6    = stats + 3;
  float* ddsum = stats + 9;   float* ddsum2 = stats + 57;
  float* xgsum = stats + 105; float* xgm6   = stats + 108;
  float* aCE = stats + 128, *cCE = stats + 176;
  float* waCD = stats + 224, *cCD = stats + 416;
  float* aDD = stats + 464, *cDD = stats + 512;
  float* aB  = stats + 560, *cB  = stats + 608;

  hipMemsetAsync(stats, 0, 512, stream);

  k_mom3<<<256, 256, 0, stream>>>(x, 65536, xsum, xm6);
  k2_coeffs1<<<1, 64, 0, stream>>>(stats, ce_w1, ce_g, ce_b, cd_wt, cd_g, cd_b,
                                   aCE, cCE, waCD, cCD);
  k3_compenv<<<1024, 256, 0, stream>>>(x, ce_w1, ce_w2, aCE, cCE, ENV);
  k4_ddconvT1<<<(4*258*258 + 127)/128, 128, 0, stream>>>(x, ENV, dd_f_wt, h1dd);
  k5_ddconv2<<<dim3(16, 32, 4), 128, 0, stream>>>(h1dd, dd_f_wc, h2dd);
  k6_ddmom<<<dim3(48, 32), 256, 0, stream>>>(h2dd, ddsum, ddsum2);
  k8_cdconv2<<<dim3(33, 65, 12), 128, 0, stream>>>(x, cd_f_wt, waCD, cCD, h2cd);
  k9_xg<<<(4*512*512 + 255)/256, 256, 0, stream>>>(h2cd, cd_f_wc, xg);
  k_mom3<<<512, 256, 0, stream>>>(xg, 262144, xgsum, xgm6);
  k11_coeffs2<<<1, 64, 0, stream>>>(stats, dd_g, dd_b, born_w1, born_g, born_b,
                                    aDD, cDD, aB, cB);
  k12_final<<<1024, 256, 0, stream>>>(h2dd, xg, dd_wt, aDD, cDD,
                                      born_w1, born_w2, aB, cB, out);
  (void)in_sizes; (void)n_in; (void)out_size;
}
